// FullAttention_37821482009267
// MI455X (gfx1250) — matrix operand order repaired and checked by reading, not compiled, not run on hardware
//
#include <hip/hip_runtime.h>
#include <hip/hip_bf16.h>
#include <math.h>

// ---------------------------------------------------------------------------
// CDNA5 (gfx1250) fused Transformer-XL attention layer.
// Heavy contractions: v_wmma_f32_16x16x32_bf16 fed from LDS tiles staged by
// the Tensor Data Mover (tensor_load_to_lds + s_wait_tensorcnt), double-
// buffered so DMA overlaps the WMMA stream.
// ---------------------------------------------------------------------------

typedef __attribute__((ext_vector_type(16))) __bf16 v16bf;
typedef __attribute__((ext_vector_type(8)))  __bf16 v8bf;
typedef __attribute__((ext_vector_type(8)))  float  v8f;
typedef __attribute__((ext_vector_type(4)))  unsigned v4u;
typedef __attribute__((ext_vector_type(8)))  int      v8i;
typedef __attribute__((ext_vector_type(4)))  int      v4i;

#define NTHREADS 256
#define BK       64   // K elements staged per LDS tile
#define PITCH    80   // bf16 elements per LDS row: 64 data + 16 pad (TDM pad)
                      // row pitch = 160B -> 16B aligned, 20-bank stride
                      // (conflict-free for the 16-lane fragment reads)

#if defined(__has_builtin)
#if __has_builtin(__builtin_amdgcn_tensor_load_to_lds) && \
    __has_builtin(__builtin_amdgcn_s_wait_tensorcnt)
#define HAVE_TDM 1
#endif
#endif
#ifndef HAVE_TDM
#define HAVE_TDM 0
#endif

#if HAVE_TDM
// Issue one TDM 2-D tile load: rows x rowlen bf16 elements from a row-major
// global matrix (leading dim stride_elems) into LDS at lds_dst, padding each
// 16 DWORDs of payload with 4 DWORDs (=> PITCH-element LDS rows).
// D# layout per CDNA5 ISA 8.3/8.4 (group0: count/lds/global/type,
// group1: data_size, pad cfg, tensor dims, tile dims, dim0 stride).
__device__ __forceinline__ void tdm_load_tile(const __bf16* gsrc,
                                              const __bf16* lds_dst,
                                              int rows, int rowlen,
                                              int stride_elems) {
  unsigned long long ga = (unsigned long long)(uintptr_t)gsrc;
  v4u g0;
  g0.x = 1u;                                    // count=1, user descriptor
  g0.y = (unsigned)(uintptr_t)lds_dst;          // flat LDS addr[31:0] = offset
  g0.z = (unsigned)ga;                          // global_addr[31:0]
  g0.w = (unsigned)((ga >> 32) & 0x01FFFFFFull) // global_addr[56:32]
         | 0x80000000u;                         // type=2 ("image")
  v8i g1;
  g1[0] = (int)(0x00010000u     // data_size=1 -> 2-byte elements
                | (1u << 20)    // pad_enable
                | (3u << 22)    // pad_interval: 16 DWORDs
                | (3u << 25));  // pad_amount: 4 DWORDs
  g1[1] = (int)(((unsigned)rowlen & 0xFFFFu) << 16);            // tdim0[15:0]
  g1[2] = (int)(((unsigned)rowlen >> 16) |
                (((unsigned)rows & 0xFFFFu) << 16));            // tdim0 hi|tdim1 lo
  g1[3] = (int)(((unsigned)rows >> 16) |
                (((unsigned)rowlen & 0xFFFFu) << 16));          // tdim1 hi|tile_dim0
  g1[4] = (int)((unsigned)rows & 0xFFFFu);                      // tile_dim1 (dim2=0)
  g1[5] = stride_elems;                                         // dim0_stride[31:0]
  g1[6] = 0;                                                    // stride hi / dim1_stride
  g1[7] = 0;
  v4i z4 = {0, 0, 0, 0};
#if defined(__clang_major__) && __clang_major__ >= 23
  v8i z8 = {0, 0, 0, 0, 0, 0, 0, 0};
  __builtin_amdgcn_tensor_load_to_lds(g0, g1, z4, z4, z8, 0);
#else
  __builtin_amdgcn_tensor_load_to_lds(g0, g1, z4, z4, 0);
#endif
}
#endif  // HAVE_TDM

// ---------------- WMMA fragment loaders from padded LDS tiles ---------------
// A (16x32 bf16, ISA layout): lane L(0-15) row M=L, K base..+7 & +16..+23;
// lanes 16-31 same rows, K offsets +8 / +24.
__device__ __forceinline__ v16bf lds_frag_a(const __bf16* base, int mloc,
                                            int s, int lane) {
  const int m    = mloc + (lane & 15);
  const int ksel = (lane >> 4) & 1;
  const __bf16* p = base + (size_t)m * PITCH + s * 40 + ksel * 8;
  v8bf lo = *(const v8bf*)p;
  v8bf hi = *(const v8bf*)(p + 16);
  v16bf f;
#pragma unroll
  for (int t = 0; t < 8; ++t) { f[t] = lo[t]; f[t + 8] = hi[t]; }
  return f;
}

// B (32x16 bf16): lane L holds column N=L%16; lanes 0-15 K base..+7 & +16..+23,
// lanes 16-31 K offsets +8 / +24 (same K order as A).
__device__ __forceinline__ v16bf lds_frag_b(const __bf16* base, int nloc,
                                            int s, int lane) {
  const int n    = nloc + (lane & 15);
  const int ksel = (lane >> 4) & 1;
  const __bf16* p = base + (size_t)n * PITCH + s * 40 + ksel * 8;
  v8bf lo = *(const v8bf*)p;
  v8bf hi = *(const v8bf*)(p + 16);
  v16bf f;
#pragma unroll
  for (int t = 0; t < 8; ++t) { f[t] = lo[t]; f[t + 8] = hi[t]; }
  return f;
}

__device__ __forceinline__ v8f wmma_bf16(v16bf a, v16bf b, v8f c) {
  return __builtin_amdgcn_wmma_f32_16x16x32_bf16(
      false, a, false, b, (short)0, c, false, false);
}

// ---------------- generic NT GEMM with fused epilogue -----------------------
// C[M,N] = epilogue( scale * (A[M,K] . Bt[N,K]^T) + bias[N] )
// act: 0 none, 1 silu.  mult: optional bf16 [M,N] elementwise gate.
// cmode: 0 -> bf16 row-major, 1 -> bf16 transposed [N,M], 2 -> f32 row-major.
// Block = 256 threads (8 waves). Block tile 128(M) x 64(N); wave tile 32x32.
// M % 128 == 0, N % 64 == 0, Kd % 64 == 0 (all shapes here comply).
__global__ __launch_bounds__(NTHREADS)
void gemm_nt_bf16(const __bf16* __restrict__ A, int lda,
                  const __bf16* __restrict__ Bt, int ldb,
                  int M, int N, int Kd,
                  float scale, const float* __restrict__ bias, int act,
                  const __bf16* __restrict__ mult, int ldmult,
                  void* __restrict__ Cout, int ldc, int cmode) {
  __shared__ alignas(16) __bf16 As[2][128 * PITCH];
  __shared__ alignas(16) __bf16 Bs[2][64 * PITCH];
  (void)M; (void)N;

  const int lane = threadIdx.x & 31;
  const int wid  = threadIdx.x >> 5;  // 0..7
  const int wm   = wid & 3;           // 4 waves along M
  const int wn   = wid >> 2;          // 2 waves along N
  const int m0   = wm * 32;           // local row of wave tile
  const int n0   = wn * 32;           // local col of wave tile
  const int mblk = blockIdx.y * 128;
  const int nblk = blockIdx.x * 64;
  const __bf16* Ag = A + (size_t)mblk * lda;
  const __bf16* Bg = Bt + (size_t)nblk * ldb;

  v8f acc00 = {}, acc01 = {}, acc10 = {}, acc11 = {};
  const int nk = Kd / BK;

#if HAVE_TDM
  if (wid == 0) {  // wave 0 drives the Tensor Data Mover
    tdm_load_tile(Ag, As[0], 128, BK, lda);
    tdm_load_tile(Bg, Bs[0], 64, BK, ldb);
  }
  for (int kt = 0; kt < nk; ++kt) {
    const int cur = kt & 1;
    if (wid == 0) {
      if (kt + 1 < nk) {  // prefetch next stage, wait for current pair
        tdm_load_tile(Ag + (size_t)(kt + 1) * BK, As[cur ^ 1], 128, BK, lda);
        tdm_load_tile(Bg + (size_t)(kt + 1) * BK, Bs[cur ^ 1], 64, BK, ldb);
        __builtin_amdgcn_s_wait_tensorcnt((short)2);
      } else {
        __builtin_amdgcn_s_wait_tensorcnt((short)0);
      }
    }
    __syncthreads();  // current tiles visible to all waves
#pragma unroll
    for (int s = 0; s < 2; ++s) {
      v16bf a0 = lds_frag_a(As[cur], m0,      s, lane);
      v16bf a1 = lds_frag_a(As[cur], m0 + 16, s, lane);
      v16bf b0 = lds_frag_b(Bs[cur], n0,      s, lane);
      v16bf b1 = lds_frag_b(Bs[cur], n0 + 16, s, lane);
      acc00 = wmma_bf16(a0, b0, acc00);
      acc01 = wmma_bf16(a0, b1, acc01);
      acc10 = wmma_bf16(a1, b0, acc10);
      acc11 = wmma_bf16(a1, b1, acc11);
    }
    __syncthreads();  // done reading before next-next prefetch overwrites
  }
#else
  // Fallback: cooperative global->LDS staging with plain loads.
  for (int kt = 0; kt < nk; ++kt) {
    const int kb = kt * BK;
    __syncthreads();
    for (int u = threadIdx.x; u < 128 * 8; u += NTHREADS) {
      const int r = u >> 3, g = u & 7;
      *(v8bf*)(As[0] + (size_t)r * PITCH + (g >> 2) * 40 + (g & 3) * 8) =
          *(const v8bf*)(Ag + (size_t)r * lda + kb + g * 8);
    }
    for (int u = threadIdx.x; u < 64 * 8; u += NTHREADS) {
      const int r = u >> 3, g = u & 7;
      *(v8bf*)(Bs[0] + (size_t)r * PITCH + (g >> 2) * 40 + (g & 3) * 8) =
          *(const v8bf*)(Bg + (size_t)r * ldb + kb + g * 8);
    }
    __syncthreads();
#pragma unroll
    for (int s = 0; s < 2; ++s) {
      v16bf a0 = lds_frag_a(As[0], m0,      s, lane);
      v16bf a1 = lds_frag_a(As[0], m0 + 16, s, lane);
      v16bf b0 = lds_frag_b(Bs[0], n0,      s, lane);
      v16bf b1 = lds_frag_b(Bs[0], n0 + 16, s, lane);
      acc00 = wmma_bf16(a0, b0, acc00);
      acc01 = wmma_bf16(a0, b1, acc01);
      acc10 = wmma_bf16(a1, b0, acc10);
      acc11 = wmma_bf16(a1, b1, acc11);
    }
  }
#endif

  // C layout: lane holds column N = lane%16; VGPR g holds row M = g (lanes
  // 0-15) or M = 8+g (lanes 16-31).
  const int cn    = lane & 15;
  const int rhalf = (lane >> 4) * 8;
#pragma unroll
  for (int tile = 0; tile < 4; ++tile) {
    const int tm = tile >> 1, tn = tile & 1;
    const v8f& c = (tile == 0) ? acc00 : (tile == 1) ? acc01
                 : (tile == 2) ? acc10 : acc11;
    const int rbase = mblk + m0 + tm * 16 + rhalf;
    const int col   = nblk + n0 + tn * 16 + cn;
    const float b = bias ? bias[col] : 0.0f;
#pragma unroll
    for (int g = 0; g < 8; ++g) {
      const int row = rbase + g;
      float v = c[g] * scale + b;
      if (act == 1) v = v * (1.0f / (1.0f + __expf(-v)));  // silu
      if (mult)     v *= (float)mult[(size_t)row * ldmult + col];
      if (cmode == 2)
        ((float*)Cout)[(size_t)row * ldc + col] = v;
      else if (cmode == 1)
        ((__bf16*)Cout)[(size_t)col * ldc + row] = (__bf16)v;
      else
        ((__bf16*)Cout)[(size_t)row * ldc + col] = (__bf16)v;
    }
  }
}

// ---------------- RMSNorm: x[row,:] * rsqrt(mean(x^2)+eps) * g -> bf16 ------
__global__ __launch_bounds__(NTHREADS)
void rmsnorm_bf16(const float* __restrict__ x, const float* __restrict__ g,
                  __bf16* __restrict__ xt, int D) {
  const int row = blockIdx.x;
  const int tid = threadIdx.x;
  const float* xr = x + (size_t)row * D;
  float ss = 0.0f;
  for (int j = tid; j < D; j += NTHREADS) { float v = xr[j]; ss += v * v; }
#pragma unroll
  for (int o = 16; o > 0; o >>= 1) ss += __shfl_xor(ss, o, 32);
  __shared__ float red[8];
  if ((tid & 31) == 0) red[tid >> 5] = ss;
  __syncthreads();
  float tot = 0.0f;
#pragma unroll
  for (int w = 0; w < 8; ++w) tot += red[w];
  const float r = rsqrtf(tot / (float)D + 1e-6f);
  for (int j = tid; j < D; j += NTHREADS)
    xt[(size_t)row * D + j] = (__bf16)(xr[j] * r * g[j]);
}

// ---------------- flipped Transformer-XL sinusoid table (bf16) --------------
__global__ __launch_bounds__(NTHREADS)
void sinusoid_bf16(__bf16* __restrict__ emb, int T, int D) {
  const int idx = blockIdx.x * NTHREADS + threadIdx.x;
  if (idx >= T * D) return;
  const int r = idx / D, c = idx % D, h = D >> 1;
  const float pos = (float)(T - 1 - r);  // flipped
  const int   c2  = (c < h) ? c : c - h;
  const float inv = __expf(-((float)(2 * c2) / (float)D) * 9.210340371976184f);
  const float pre = pos * inv;
  emb[idx] = (__bf16)((c < h) ? __sinf(pre) : __cosf(pre));
}

// ---------------- f32 [R,C] -> bf16 transposed [C,R] ------------------------
__global__ __launch_bounds__(NTHREADS)
void transpose_f32_to_bf16(const float* __restrict__ in,
                           __bf16* __restrict__ out, int R, int C) {
  const int idx = blockIdx.x * NTHREADS + threadIdx.x;
  if (idx >= R * C) return;
  const int r = idx / C, c = idx % C;
  out[(size_t)c * R + r] = (__bf16)in[idx];
}

// ---------------- rel_shift + causal mask + row softmax -> bf16 probs -------
// s[i,j] = ac[i,j] + bd[i, j-i+T-1] for j<=i else -inf. One block per row.
__global__ __launch_bounds__(NTHREADS)
void relshift_softmax(const float* __restrict__ ac,
                      const float* __restrict__ bd,
                      __bf16* __restrict__ w, int T) {
  const int i   = blockIdx.x;
  const int tid = threadIdx.x;
  float sv[16];  // T / NTHREADS = 16 columns per thread
  float m = -1e30f;
  int t = 0;
  for (int j = tid; j < T; j += NTHREADS, ++t) {
    float s = -1e30f;
    if (j <= i)
      s = ac[(size_t)i * T + j] + bd[(size_t)i * T + (j - i + T - 1)];
    sv[t] = s;
    m = fmaxf(m, s);
  }
#pragma unroll
  for (int o = 16; o > 0; o >>= 1) m = fmaxf(m, __shfl_xor(m, o, 32));
  __shared__ float smax[8], ssum[8];
  if ((tid & 31) == 0) smax[tid >> 5] = m;
  __syncthreads();
  float M = -1e30f;
#pragma unroll
  for (int v = 0; v < 8; ++v) M = fmaxf(M, smax[v]);
  float z = 0.0f;
#pragma unroll
  for (int u = 0; u < 16; ++u) { sv[u] = __expf(sv[u] - M); z += sv[u]; }
#pragma unroll
  for (int o = 16; o > 0; o >>= 1) z += __shfl_xor(z, o, 32);
  if ((tid & 31) == 0) ssum[tid >> 5] = z;
  __syncthreads();
  float Z = 0.0f;
#pragma unroll
  for (int v = 0; v < 8; ++v) Z += ssum[v];
  const float invZ = 1.0f / Z;
  t = 0;
  for (int j = tid; j < T; j += NTHREADS, ++t)
    w[(size_t)i * T + j] = (__bf16)(sv[t] * invZ);  // 0 above diagonal
}

// ---------------------------------------------------------------------------
static inline size_t align256(size_t x) { return (x + 255) & ~(size_t)255; }

extern "C" void kernel_launch(void* const* d_in, const int* in_sizes, int n_in,
                              void* d_out, int out_size, void* d_ws,
                              size_t ws_size, hipStream_t stream) {
  (void)in_sizes; (void)n_in; (void)out_size; (void)ws_size;
  const int B = 2, T = 4096, D = 1024, K = 128, V = 2048;
  const float inv_tau = 1.0f / sqrtf((float)K);

  const float* x    = (const float*)d_in[0];
  const float* g_ln = (const float*)d_in[1];
  const float* W_q  = (const float*)d_in[2];
  const float* W_k  = (const float*)d_in[3];
  const float* W_v  = (const float*)d_in[4];
  const float* W_g  = (const float*)d_in[5];
  const float* W_r  = (const float*)d_in[6];
  const float* xl_u = (const float*)d_in[7];
  const float* xl_v = (const float*)d_in[8];
  const float* W_o  = (const float*)d_in[9];
  float* out = (float*)d_out;

  // Workspace carve-up (~310 MB total).
  char* ws = (char*)d_ws;
  size_t off = 0;
  auto take = [&](size_t bytes) -> char* {
    char* p = ws + off; off = align256(off + bytes); return p;
  };
  __bf16* xt   = (__bf16*)take((size_t)B * T * D * 2);  // x_tilde bf16
  __bf16* WqT  = (__bf16*)take((size_t)K * D * 2);
  __bf16* WkT  = (__bf16*)take((size_t)K * D * 2);
  __bf16* WvT  = (__bf16*)take((size_t)V * D * 2);
  __bf16* WgT  = (__bf16*)take((size_t)V * D * 2);
  __bf16* WrT  = (__bf16*)take((size_t)K * D * 2);
  __bf16* WoT  = (__bf16*)take((size_t)D * V * 2);
  __bf16* qu   = (__bf16*)take((size_t)B * T * K * 2);  // q/tau + xl_u
  __bf16* qv   = (__bf16*)take((size_t)B * T * K * 2);  // q/tau + xl_v
  __bf16* kb   = (__bf16*)take((size_t)B * T * K * 2);
  __bf16* vT   = (__bf16*)take((size_t)B * V * T * 2);  // v transposed [B,V,T]
  __bf16* gate = (__bf16*)take((size_t)B * T * V * 2);  // silu(x~ Wg)
  __bf16* emb  = (__bf16*)take((size_t)T * D * 2);
  __bf16* rbf  = (__bf16*)take((size_t)T * K * 2);
  float*  acb  = (float*) take((size_t)T * T * 4);      // per-batch, reused
  float*  bdb  = (float*) take((size_t)T * T * 4);      // per-batch, reused
  __bf16* wbuf = (__bf16*)take((size_t)T * T * 2);      // per-batch, reused
  __bf16* wvg  = (__bf16*)take((size_t)B * T * V * 2);  // (w@v)*gate

  const dim3 blk(NTHREADS);
  auto ggrid = [](int M, int N) { return dim3(N / 64, M / 128); };
  auto egrid = [](size_t n) {
    return dim3((unsigned)((n + NTHREADS - 1) / NTHREADS));
  };

  // 1) weight transposes (once-per-call, ~13 MB)
  transpose_f32_to_bf16<<<egrid((size_t)D * K), blk, 0, stream>>>(W_q, WqT, D, K);
  transpose_f32_to_bf16<<<egrid((size_t)D * K), blk, 0, stream>>>(W_k, WkT, D, K);
  transpose_f32_to_bf16<<<egrid((size_t)D * V), blk, 0, stream>>>(W_v, WvT, D, V);
  transpose_f32_to_bf16<<<egrid((size_t)D * V), blk, 0, stream>>>(W_g, WgT, D, V);
  transpose_f32_to_bf16<<<egrid((size_t)D * K), blk, 0, stream>>>(W_r, WrT, D, K);
  transpose_f32_to_bf16<<<egrid((size_t)V * D), blk, 0, stream>>>(W_o, WoT, V, D);

  // 2) RMSNorm + sinusoid table
  rmsnorm_bf16<<<dim3(B * T), blk, 0, stream>>>(x, g_ln, xt, D);
  sinusoid_bf16<<<egrid((size_t)T * D), blk, 0, stream>>>(emb, T, D);

  // 3) projections (all WMMA NT GEMMs)
  gemm_nt_bf16<<<ggrid(B * T, K), blk, 0, stream>>>(xt, D, WqT, D, B * T, K, D,
      inv_tau, xl_u, 0, nullptr, 0, qu, K, 0);
  gemm_nt_bf16<<<ggrid(B * T, K), blk, 0, stream>>>(xt, D, WqT, D, B * T, K, D,
      inv_tau, xl_v, 0, nullptr, 0, qv, K, 0);
  gemm_nt_bf16<<<ggrid(B * T, K), blk, 0, stream>>>(xt, D, WkT, D, B * T, K, D,
      1.0f, nullptr, 0, nullptr, 0, kb, K, 0);
  for (int b = 0; b < B; ++b)  // v, stored transposed [V,T] per batch
    gemm_nt_bf16<<<ggrid(T, V), blk, 0, stream>>>(xt + (size_t)b * T * D, D,
        WvT, D, T, V, D, 1.0f, nullptr, 0, nullptr, 0,
        vT + (size_t)b * V * T, T, 1);
  gemm_nt_bf16<<<ggrid(B * T, V), blk, 0, stream>>>(xt, D, WgT, D, B * T, V, D,
      1.0f, nullptr, 1 /*silu*/, nullptr, 0, gate, V, 0);
  gemm_nt_bf16<<<ggrid(T, K), blk, 0, stream>>>(emb, D, WrT, D, T, K, D,
      1.0f, nullptr, 0, nullptr, 0, rbf, K, 0);

  // 4) per-batch attention: AC, BDraw, rel-shift softmax, (w @ v) * gate
  for (int b = 0; b < B; ++b) {
    const __bf16* qub = qu + (size_t)b * T * K;
    const __bf16* qvb = qv + (size_t)b * T * K;
    const __bf16* kbb = kb + (size_t)b * T * K;
    gemm_nt_bf16<<<ggrid(T, T), blk, 0, stream>>>(qub, K, kbb, K, T, T, K,
        1.0f, nullptr, 0, nullptr, 0, acb, T, 2);
    gemm_nt_bf16<<<ggrid(T, T), blk, 0, stream>>>(qvb, K, rbf, K, T, T, K,
        1.0f, nullptr, 0, nullptr, 0, bdb, T, 2);
    relshift_softmax<<<dim3(T), blk, 0, stream>>>(acb, bdb, wbuf, T);
    gemm_nt_bf16<<<ggrid(T, V), blk, 0, stream>>>(wbuf, T,
        vT + (size_t)b * V * T, T, T, V, T, 1.0f, nullptr, 0,
        gate + (size_t)b * T * V, V, wvg + (size_t)b * T * V, V, 0);
  }

  // 5) output projection -> f32
  gemm_nt_bf16<<<ggrid(B * T, D), blk, 0, stream>>>(wvg, V, WoT, V,
      B * T, D, V, 1.0f, nullptr, 0, nullptr, 0, out, D, 2);
}
